// ScalarGatedSelfAttention_82523501625520
// MI455X (gfx1250) — matrix operand order repaired and checked by reading, not compiled, not run on hardware
//
#include <hip/hip_runtime.h>

// ---------------------------------------------------------------------------
// ScalarGatedSelfAttention for MI455X (gfx1250), wave32 + WMMA bf16.
//   B=16, H=W=64, C=512, Ci=64.  Memory-bound (~270MB HBM traffic, ~13 GFLOP),
//   so: bf16 WMMA for all 3 GEMMs, f32 accumulate, LN fused into epilogues.
// Workspace layout (needs ~34 MB):
//   [0..64K)      WfT   bf16 [64][512]   (Wf transposed,  N-major)
//   [64K..128K)   WhT   bf16 [64][512]
//   [128K..192K)  WfghT bf16 [512][64]
//   [192K..+16M)  fg = LN(x@Wf) fp32 [B,H,W,64]
//   [+16M..+32M)  hm = LN(x@Wh) fp32 [B,H,W,64]
// ---------------------------------------------------------------------------

typedef unsigned short u16;
typedef unsigned int   u32;
typedef __attribute__((ext_vector_type(16))) __bf16 v16bf;
typedef __attribute__((ext_vector_type(8)))  float  v8f;

union Frag {            // one WMMA A/B operand: 16 bf16 = 32 bytes
    uint4 q[2];
    v16bf v;
};

__device__ __forceinline__ u16 f2bf(float f) {          // RNE fp32 -> bf16
    u32 u = __float_as_uint(f);
    u32 r = u + 0x7FFFu + ((u >> 16) & 1u);
    return (u16)(r >> 16);
}

// ---------------------------------------------------------------------------
// prep: transpose + downconvert the 3 weight matrices (32768 elements each)
// ---------------------------------------------------------------------------
__global__ __launch_bounds__(256) void prep_weights(
    const float* __restrict__ Wf, const float* __restrict__ Wh,
    const float* __restrict__ Wfgh,
    u16* __restrict__ WfT, u16* __restrict__ WhT, u16* __restrict__ WfghT)
{
    int i = blockIdx.x * 256 + threadIdx.x;            // 0 .. 32767
    if (i < 512 * 64) {
        int k = i >> 6, n = i & 63;                    // Wf[k][n], k<512, n<64
        WfT[(size_t)n * 512 + k] = f2bf(Wf[i]);
        WhT[(size_t)n * 512 + k] = f2bf(Wh[i]);
        int kk = i >> 9, nn = i & 511;                 // Wfgh[kk][nn], kk<64
        WfghT[(size_t)nn * 64 + kk] = f2bf(Wfgh[i]);
    }
}

// ---------------------------------------------------------------------------
// pass1: per 64-token tile: bf16 GEMM x@Wf and x@Wh + fused LayerNorm(Ci=64)
// ---------------------------------------------------------------------------
union SMem1 {
    u16   xt[64][264];        // x tile bf16, K-chunk of 256 (+8 pad)  33.8 KB
    float res[2][64][68];     // f / h pre-LN results                  34.8 KB
};

__global__ __launch_bounds__(256) void pass1_gemm_ln(
    const float* __restrict__ x,
    const u16* __restrict__ WfT, const u16* __restrict__ WhT,
    const float* __restrict__ gf, const float* __restrict__ bfv,
    const float* __restrict__ gh, const float* __restrict__ bhv,
    float* __restrict__ fg, float* __restrict__ hm)
{
    __shared__ SMem1 sm;
    const int tid  = threadIdx.x;
    const size_t tok0 = (size_t)blockIdx.x * 64;
    const int wave = tid >> 5, lane = tid & 31;
    const int mg   = wave & 3;           // M group (16 tokens)
    const int sel  = wave >> 2;          // 0 = f branch, 1 = h branch
    const u16* BT  = sel ? WhT : WfT;
    const int mrow = mg * 16 + (lane & 15);
    const int koA  = (lane >> 4) << 3;   // A: upper half-wave -> K+8
    const int koB  = (lane >> 4) << 3;   // B: upper half-wave -> K+8
    const int nlan = lane & 15;

    const v8f vzero = {0.f,0.f,0.f,0.f,0.f,0.f,0.f,0.f};
    v8f acc[4];
#pragma unroll
    for (int t = 0; t < 4; ++t) acc[t] = vzero;

    for (int kc = 0; kc < 2; ++kc) {                   // two K chunks of 256
        // ---- load 64x256 fp32 of x, convert to bf16 in LDS ----
#pragma unroll
        for (int it = 0; it < 16; ++it) {
            int idx = it * 256 + tid;                  // 0..4095 float4s
            int r = idx >> 6, c4 = (idx & 63) << 2;
            float4 v = *(const float4*)(x + (tok0 + r) * 512 + kc * 256 + c4);
            uint2 pk;
            pk.x = (u32)f2bf(v.x) | ((u32)f2bf(v.y) << 16);
            pk.y = (u32)f2bf(v.z) | ((u32)f2bf(v.w) << 16);
            *(uint2*)&sm.xt[r][c4] = pk;
        }
        __syncthreads();
        // ---- WMMA: 8 K-steps x 4 N-tiles ----
        for (int k0 = 0; k0 < 256; k0 += 32) {
            Frag a;
            const u16* ap = &sm.xt[mrow][k0 + koA];
            a.q[0] = *(const uint4*)ap;                // K = ko .. ko+7
            a.q[1] = *(const uint4*)(ap + 16);         // K = ko+16 .. ko+23
#pragma unroll
            for (int t = 0; t < 4; ++t) {
                int n = t * 16 + nlan;
                Frag b;
                const u16* bp = BT + (size_t)n * 512 + kc * 256 + k0 + koB;
                b.q[0] = *(const uint4*)bp;
                b.q[1] = *(const uint4*)(bp + 16);
                acc[t] = __builtin_amdgcn_wmma_f32_16x16x32_bf16(
                    false, a.v, false, b.v, (short)0, acc[t], false, false);
            }
        }
        __syncthreads();                               // xt dead -> reuse as res
    }

    // ---- scatter accumulators to LDS (C layout: lane-half selects M+8) ----
#pragma unroll
    for (int t = 0; t < 4; ++t)
#pragma unroll
        for (int r = 0; r < 8; ++r) {
            int M = mg * 16 + r + ((lane >> 4) << 3);
            sm.res[sel][M][t * 16 + nlan] = acc[t][r];
        }
    __syncthreads();

    // ---- fused LayerNorm over Ci=64, write fp32 intermediates ----
    if (tid < 128) {
        int s = tid >> 6, t = tid & 63;
        const float* row = sm.res[s][t];
        float su = 0.f, sq = 0.f;
        for (int d = 0; d < 64; ++d) { float v = row[d]; su += v; sq += v * v; }
        float mean = su * (1.f / 64.f);
        float rstd = rsqrtf(sq * (1.f / 64.f) - mean * mean + 1e-3f);
        const float* ga = s ? gh  : gf;
        const float* be = s ? bhv : bfv;
        float* dst = (s ? hm : fg) + (tok0 + t) * 64;
        for (int d = 0; d < 64; d += 4) {
            float4 o;
            o.x = (row[d+0] - mean) * rstd * ga[d+0] + be[d+0];
            o.y = (row[d+1] - mean) * rstd * ga[d+1] + be[d+1];
            o.z = (row[d+2] - mean) * rstd * ga[d+2] + be[d+2];
            o.w = (row[d+3] - mean) * rstd * ga[d+3] + be[d+3];
            *(float4*)(dst + d) = o;
        }
    }
}

// ---------------------------------------------------------------------------
// pass2: per (b,w): softmax over H, t = attn*h, t@Wfgh (WMMA), LN(C=512),
//        residual + scale.  LN uses shfl_xor reductions over accumulators.
// ---------------------------------------------------------------------------
__global__ __launch_bounds__(256) void pass2_attn_out(
    const float* __restrict__ fg, const float* __restrict__ hm,
    const u16* __restrict__ WfghT,
    const float* __restrict__ gamma, const float* __restrict__ beta,
    const float* __restrict__ scaleArr,
    const float* __restrict__ x, float* __restrict__ out)
{
    __shared__ float fgA[64][64];     // fg[b, w, h, d]  ("f")        16 KB
    __shared__ float fgB[64][64];     // fg[b, h, w, d]  ("g")        16 KB
    __shared__ float hB [64][64];     // hm[b, h, w, d]               16 KB
    __shared__ u16   tA [64][72];     // bf16 attn*h, A of GEMM3       9 KB
    __shared__ float red [64][2][2];  // per-row partial sum/sumsq
    __shared__ float red2[64][2];     // mean / rstd per row

    const int tid = threadIdx.x;
    const int bi  = blockIdx.x >> 6;
    const int w   = blockIdx.x & 63;

    // ---- load the three 64x64 fp32 slabs ----
#pragma unroll
    for (int it = 0; it < 4; ++it) {
        int idx = it * 256 + tid;                       // 0..1023 float4s
        ((float4*)fgA)[idx] =
            ((const float4*)(fg + (((size_t)bi * 64 + w) * 64) * 64))[idx];
        int h = idx >> 4, c4 = (idx & 15) << 2;
        size_t roff = (((size_t)bi * 64 + h) * 64 + w) * 64 + c4;
        *(float4*)&fgB[h][c4] = *(const float4*)(fg + roff);
        *(float4*)&hB [h][c4] = *(const float4*)(hm + roff);
    }
    __syncthreads();

    // ---- softmax over H (axis=1) per channel d, then t = attn * h ----
    if (tid < 64) {
        int d = tid;
        float mx = -3.4e38f;
        for (int h = 0; h < 64; ++h)
            mx = fmaxf(mx, fgA[h][d] * fgB[h][d]);
        float sum = 0.f;
        for (int h = 0; h < 64; ++h)
            sum += __expf(fgA[h][d] * fgB[h][d] - mx);
        float inv = 1.f / sum;
        for (int h = 0; h < 64; ++h) {
            float t = __expf(fgA[h][d] * fgB[h][d] - mx) * inv * hB[h][d];
            tA[h][d] = f2bf(t);
        }
    }
    __syncthreads();

    // ---- GEMM3: [64 x 64] @ [64 x 512], 8 waves x 16 N-tiles x 2 K-steps ----
    const int wave = tid >> 5, lane = tid & 31;
    const int mg = wave >> 1, nh = wave & 1;
    const int mrow = mg * 16 + (lane & 15);
    const int koA  = (lane >> 4) << 3;
    const int koB  = (lane >> 4) << 3;
    const int nlan = lane & 15;

    const v8f vzero = {0.f,0.f,0.f,0.f,0.f,0.f,0.f,0.f};
    v8f acc[16];
#pragma unroll
    for (int j = 0; j < 16; ++j) acc[j] = vzero;

#pragma unroll
    for (int kk = 0; kk < 2; ++kk) {
        int k0 = kk * 32;
        Frag a;
        const u16* ap = &tA[mrow][k0 + koA];
        a.q[0] = *(const uint4*)ap;
        a.q[1] = *(const uint4*)(ap + 16);
#pragma unroll
        for (int j = 0; j < 16; ++j) {
            int n = nh * 256 + j * 16 + nlan;
            Frag b;
            const u16* bp = WfghT + (size_t)n * 64 + k0 + koB;
            b.q[0] = *(const uint4*)bp;
            b.q[1] = *(const uint4*)(bp + 16);
            acc[j] = __builtin_amdgcn_wmma_f32_16x16x32_bf16(
                false, a.v, false, b.v, (short)0, acc[j], false, false);
        }
    }

    // ---- LN over C=512: per-row sum/sumsq via shfl within 16-lane halves ----
#pragma unroll
    for (int r = 0; r < 8; ++r) {
        float s = 0.f, q = 0.f;
#pragma unroll
        for (int j = 0; j < 16; ++j) { float v = acc[j][r]; s += v; q += v * v; }
        for (int m = 1; m < 16; m <<= 1) {
            s += __shfl_xor(s, m, 32);
            q += __shfl_xor(q, m, 32);
        }
        if (nlan == 0) {
            int M = mg * 16 + r + ((lane >> 4) << 3);
            red[M][nh][0] = s;
            red[M][nh][1] = q;
        }
    }
    __syncthreads();
    if (tid < 64) {
        float s = red[tid][0][0] + red[tid][1][0];
        float q = red[tid][0][1] + red[tid][1][1];
        float mean = s * (1.f / 512.f);
        float var  = q * (1.f / 512.f) - mean * mean;
        red2[tid][0] = mean;
        red2[tid][1] = rsqrtf(var + 1e-3f);
    }
    __syncthreads();

    // ---- epilogue: (y-mean)*rstd*gamma+beta, *scale, + x residual ----
    const float sc = scaleArr[0];
#pragma unroll
    for (int r = 0; r < 8; ++r) {
        int M = mg * 16 + r + ((lane >> 4) << 3);
        float mean = red2[M][0], rstd = red2[M][1];
        size_t rowoff = (((size_t)bi * 64 + M) * 64 + w) * 512;
#pragma unroll 4
        for (int j = 0; j < 16; ++j) {
            int n = nh * 256 + j * 16 + nlan;
            float v = (acc[j][r] - mean) * rstd * gamma[n] + beta[n];
            out[rowoff + n] = v * sc + x[rowoff + n];
        }
    }
}

// ---------------------------------------------------------------------------
extern "C" void kernel_launch(void* const* d_in, const int* in_sizes, int n_in,
                              void* d_out, int out_size, void* d_ws, size_t ws_size,
                              hipStream_t stream) {
    const float* x         = (const float*)d_in[0];
    const float* Wf        = (const float*)d_in[1];
    const float* Wh        = (const float*)d_in[2];
    const float* Wfgh      = (const float*)d_in[3];
    const float* gamma_f   = (const float*)d_in[4];
    const float* beta_f    = (const float*)d_in[5];
    const float* gamma_h   = (const float*)d_in[6];
    const float* beta_h    = (const float*)d_in[7];
    const float* gamma_fgh = (const float*)d_in[8];
    const float* beta_fgh  = (const float*)d_in[9];
    const float* scale     = (const float*)d_in[10];
    float* out = (float*)d_out;

    char* wsb = (char*)d_ws;
    u16*   WfT   = (u16*)(wsb);
    u16*   WhT   = (u16*)(wsb + 65536);
    u16*   WfghT = (u16*)(wsb + 131072);
    float* fg    = (float*)(wsb + 196608);
    float* hm    = (float*)(wsb + 196608 + 16777216ull);
    // requires ws_size >= ~33.8 MB
    (void)in_sizes; (void)n_in; (void)out_size; (void)ws_size;

    prep_weights<<<128, 256, 0, stream>>>(Wf, Wh, Wfgh, WfT, WhT, WfghT);
    pass1_gemm_ln<<<1024, 256, 0, stream>>>(x, WfT, WhT, gamma_f, beta_f,
                                            gamma_h, beta_h, fg, hm);
    pass2_attn_out<<<1024, 256, 0, stream>>>(fg, hm, WfghT, gamma_fgh, beta_fgh,
                                             scale, x, out);
}
